// Policy_Controller_54236847014219
// MI455X (gfx1250) — matrix operand order repaired and checked by reading, not compiled, not run on hardware
//
#include <hip/hip_runtime.h>
#include <hip/hip_bf16.h>

// ---------------------------------------------------------------------------
// Policy controller (LSTM scan, P=256 steps x 4 substeps) for MI455X/gfx1250.
// Single persistent workgroup (sequential dependency chain); hidden GEMV
// mapped onto v_wmma_f32_16x16x32_f16 (f16 in / f32 accumulate), weights
// pre-converted to f16 in d_ws, everything else (gates, heads, threefry RNG,
// Gumbel-max categorical) done in VALU/SALU on the same workgroup.
// ---------------------------------------------------------------------------

typedef __attribute__((ext_vector_type(16))) _Float16 v16h;
typedef __attribute__((ext_vector_type(8)))  float    v8f;

#define P_STEPS 256
#define H_DIM   512
#define E_DIM   32
#define T_OPS   16
#define M_MAGS  10
#define G4      2048   // 4*H

union F16Frag {
    v16h  v;
    uint4 q[2];
};

// ---- JAX threefry-2x32 (20 rounds) ----------------------------------------
__device__ __forceinline__ void tf2x32(unsigned k0, unsigned k1,
                                       unsigned x0, unsigned x1,
                                       unsigned* o0, unsigned* o1) {
    unsigned ks0 = k0, ks1 = k1, ks2 = k0 ^ k1 ^ 0x1BD11BDAu;
    unsigned ks[3] = {ks0, ks1, ks2};
    const int rotA[4] = {13, 15, 26, 6};
    const int rotB[4] = {17, 29, 16, 24};
    x0 += ks[0]; x1 += ks[1];
    #pragma unroll
    for (int i = 0; i < 5; ++i) {
        const int* r = (i & 1) ? rotB : rotA;
        #pragma unroll
        for (int j = 0; j < 4; ++j) {
            x0 += x1;
            x1 = (x1 << r[j]) | (x1 >> (32 - r[j]));
            x1 ^= x0;
        }
        x0 += ks[(i + 1) % 3];
        x1 += ks[(i + 2) % 3] + (unsigned)(i + 1);
    }
    *o0 = x0; *o1 = x1;
}

__device__ __forceinline__ float bits_to_uniform(unsigned b) {
    // jax uniform(minval=tiny, maxval=1)
    const float TINY = 1.1754943508222875e-38f;
    float f = __uint_as_float((b >> 9) | 0x3f800000u) - 1.0f;
    f = f * (1.0f - TINY) + TINY;
    return fmaxf(f, TINY);
}

__device__ __forceinline__ float sigmoidf(float x) {
    return 1.0f / (1.0f + expf(-x));
}

// ---- weight f32 -> f16 prep ------------------------------------------------
__global__ void f32_to_f16_kernel(const float* __restrict__ src,
                                  _Float16* __restrict__ dst, int n) {
    int i = blockIdx.x * blockDim.x + threadIdx.x;
    if (i < n) dst[i] = (_Float16)src[i];
}

// ---- main persistent controller kernel ------------------------------------
__global__ __launch_bounds__(512)
void controller_kernel(const int*   __restrict__ xin,
                       const float* __restrict__ h0,
                       const float* __restrict__ c0,
                       const float* __restrict__ emb,
                       const float* __restrict__ b_ih,
                       const float* __restrict__ b_hh,
                       const float* __restrict__ W1,
                       const float* __restrict__ b1,
                       const float* __restrict__ W2,
                       const float* __restrict__ b2,
                       const _Float16* __restrict__ Wih16,
                       const _Float16* __restrict__ Whh16,
                       float* __restrict__ out) {
    __shared__ float gbuf[G4];                       // gate pre-activations
    __shared__ float bsum[G4];                       // b_ih + b_hh
    __shared__ float h32s[H_DIM];
    __shared__ float cst[H_DIM];
    __shared__ __align__(16) _Float16 h16s[H_DIM];
    __shared__ __align__(16) _Float16 inp16[E_DIM];
    __shared__ float logits_s[T_OPS];
    __shared__ int samp_s;
    __shared__ unsigned subkey0[4], subkey1[4];
    __shared__ float lp_acc_s, ent_acc_s;

    const int tid  = threadIdx.x;
    const int lane = tid & 31;
    const int w    = tid >> 5;          // wave id, 0..15
    const int gsel = lane >> 4;         // lane half-group (0 or 1)
    const int mrow = lane & 15;         // A-matrix row within tile

    // ---- init state ----
    for (int i = tid; i < G4; i += 512) bsum[i] = b_ih[i] + b_hh[i];
    if (tid < H_DIM) {
        cst[tid]  = c0[tid];
        float hv  = h0[tid];
        h32s[tid] = hv;
        h16s[tid] = (_Float16)hv;
    }
    if (tid < E_DIM) {
        int tok0 = xin[0];
        inp16[tid] = (_Float16)emb[tok0 * E_DIM + tid];
    }
    __syncthreads();

    for (int p = 0; p < P_STEPS; ++p) {
        for (int sub = 0; sub < 4; ++sub) {
            // ============ gate pre-activations via WMMA ============
            // Each wave owns 8 output tiles of 16 rows: rows [w*128, w*128+128)
            v8f acc[8];

            // W_ih @ inp : single K=32 chunk
            {
                F16Frag bfrag;
                const _Float16* bp = &inp16[gsel * 16];
                bfrag.q[0] = *(const uint4*)(bp);
                bfrag.q[1] = *(const uint4*)(bp + 8);
                #pragma unroll
                for (int t = 0; t < 8; ++t) {
                    const int row = w * 128 + t * 16 + mrow;
                    F16Frag afrag;
                    const _Float16* ap = Wih16 + row * E_DIM + gsel * 16;
                    afrag.q[0] = *(const uint4*)(ap);
                    afrag.q[1] = *(const uint4*)(ap + 8);
                    v8f z = {};
                    acc[t] = __builtin_amdgcn_wmma_f32_16x16x32_f16(
                        false, afrag.v, false, bfrag.v, (short)0, z, false, false);
                }
            }

            // W_hh @ h : K = 512 in 16 chunks of 32
            #pragma unroll 2
            for (int kc = 0; kc < 16; ++kc) {
                F16Frag bfrag;
                const _Float16* hp = &h16s[kc * 32 + gsel * 16];
                bfrag.q[0] = *(const uint4*)(hp);
                bfrag.q[1] = *(const uint4*)(hp + 8);
                #pragma unroll
                for (int t = 0; t < 8; ++t) {
                    const int row = w * 128 + t * 16 + mrow;
                    F16Frag afrag;
                    const _Float16* ap = Whh16 + row * H_DIM + kc * 32 + gsel * 16;
                    afrag.q[0] = *(const uint4*)(ap);
                    afrag.q[1] = *(const uint4*)(ap + 8);
                    acc[t] = __builtin_amdgcn_wmma_f32_16x16x32_f16(
                        false, afrag.v, false, bfrag.v, (short)0, acc[t], false, false);
                }
            }

            // Extract column 0 (B was replicated: every column identical).
            // D layout: lane N=lane&15, VGPR r holds M = (lane>>4)*8 + r.
            if (mrow == 0) {
                #pragma unroll
                for (int t = 0; t < 8; ++t) {
                    const int ob = w * 128 + t * 16 + gsel * 8;
                    #pragma unroll
                    for (int r = 0; r < 8; ++r) {
                        const int gi = ob + r;
                        gbuf[gi] = acc[t][r] + bsum[gi];
                    }
                }
            }
            __syncthreads();

            // ============ LSTM cell update (i,f,g,o) ============
            {
                const int u = tid;  // blockDim == 512 == H
                float gi_ = gbuf[u];
                float gf_ = gbuf[H_DIM + u];
                float gg_ = gbuf[2 * H_DIM + u];
                float go_ = gbuf[3 * H_DIM + u];
                float cn  = sigmoidf(gf_) * cst[u] + sigmoidf(gi_) * tanhf(gg_);
                float hn  = sigmoidf(go_) * tanhf(cn);
                cst[u]  = cn;
                h32s[u] = hn;
                h16s[u] = (_Float16)hn;
            }
            __syncthreads();

            // ============ head logits (K x 512 GEMV) ============
            const int    K  = (sub & 1) ? M_MAGS : T_OPS;
            const float* Wf = (sub & 1) ? W2 : W1;
            const float* bf = (sub & 1) ? b2 : b1;
            if (w < K) {
                float s = 0.0f;
                #pragma unroll
                for (int j = lane; j < H_DIM; j += 32)
                    s += h32s[j] * Wf[w * H_DIM + j];
                #pragma unroll
                for (int off = 16; off > 0; off >>= 1)
                    s += __shfl_down(s, off, 32);
                if (lane == 0) logits_s[w] = s + bf[w];
            }
            __syncthreads();

            // ============ sampling / logprob / entropy (thread 0) ============
            if (tid == 0) {
                if (sub == 0) {
                    // step key: keys = split(key(42), 256); kp = keys[p]
                    unsigned a0, a1, b0, b1_;
                    unsigned kp0, kp1;
                    if (p < 128) {
                        tf2x32(0u, 42u, (unsigned)(2 * p),     (unsigned)(2 * p + 256), &a0, &a1);
                        tf2x32(0u, 42u, (unsigned)(2 * p + 1), (unsigned)(2 * p + 257), &b0, &b1_);
                        kp0 = a0; kp1 = b0;
                    } else {
                        const int i0 = 2 * p - 256, i1 = 2 * p - 255;
                        tf2x32(0u, 42u, (unsigned)i0, (unsigned)(i0 + 256), &a0, &a1);
                        tf2x32(0u, 42u, (unsigned)i1, (unsigned)(i1 + 256), &b0, &b1_);
                        kp0 = a1; kp1 = b1_;
                    }
                    // split(kp, 4): counts iota(8) -> halves [0..3],[4..7]
                    unsigned y0[4], y1[4];
                    #pragma unroll
                    for (int i2 = 0; i2 < 4; ++i2)
                        tf2x32(kp0, kp1, (unsigned)i2, (unsigned)(i2 + 4), &y0[i2], &y1[i2]);
                    subkey0[0] = y0[0]; subkey1[0] = y0[1];
                    subkey0[1] = y0[2]; subkey1[1] = y0[3];
                    subkey0[2] = y1[0]; subkey1[2] = y1[1];
                    subkey0[3] = y1[2]; subkey1[3] = y1[3];
                    lp_acc_s = 0.0f; ent_acc_s = 0.0f;
                }
                // uniform bits -> gumbel
                const unsigned sk0 = subkey0[sub], sk1 = subkey1[sub];
                const int half = K / 2;   // K = 16 or 10, both even
                float gumb[T_OPS];
                for (int i2 = 0; i2 < half; ++i2) {
                    unsigned o0, o1;
                    tf2x32(sk0, sk1, (unsigned)i2, (unsigned)(i2 + half), &o0, &o1);
                    gumb[i2]        = -logf(-logf(bits_to_uniform(o0)));
                    gumb[i2 + half] = -logf(-logf(bits_to_uniform(o1)));
                }
                // Gumbel-max argmax (first max, like jnp.argmax)
                int   samp = 0;
                float best = -3.4e38f, lmax = -3.4e38f;
                for (int i2 = 0; i2 < K; ++i2) {
                    float z = logits_s[i2] + gumb[i2];
                    if (z > best) { best = z; samp = i2; }
                    lmax = fmaxf(lmax, logits_s[i2]);
                }
                float se = 0.0f;
                for (int i2 = 0; i2 < K; ++i2) se += expf(logits_s[i2] - lmax);
                float lse = logf(se);
                float lp  = logits_s[samp] - lmax - lse;
                float ent = 0.0f;
                for (int i2 = 0; i2 < K; ++i2) {
                    float logp = logits_s[i2] - lmax - lse;
                    ent -= expf(logp) * logp;
                }
                lp_acc_s  += lp;
                ent_acc_s += ent;
                out[2 * P_STEPS + 4 * p + sub] = (float)samp;
                if (sub == 3) {
                    out[p]           = lp_acc_s;
                    out[P_STEPS + p] = ent_acc_s;
                }
                samp_s = ((sub & 1) ? (1 + T_OPS) : 1) + samp;
            }
            __syncthreads();

            // next embedding input
            if (tid < E_DIM) {
                inp16[tid] = (_Float16)emb[samp_s * E_DIM + tid];
            }
            __syncthreads();
        }
    }
}

extern "C" void kernel_launch(void* const* d_in, const int* in_sizes, int n_in,
                              void* d_out, int out_size, void* d_ws, size_t ws_size,
                              hipStream_t stream) {
    (void)in_sizes; (void)n_in; (void)out_size; (void)ws_size;

    const int*   x    = (const int*)  d_in[0];
    const float* h0   = (const float*)d_in[1];
    const float* c0   = (const float*)d_in[2];
    const float* emb  = (const float*)d_in[3];
    const float* W_ih = (const float*)d_in[4];
    const float* b_ih = (const float*)d_in[5];
    const float* W_hh = (const float*)d_in[6];
    const float* b_hh = (const float*)d_in[7];
    const float* W1   = (const float*)d_in[8];
    const float* b1   = (const float*)d_in[9];
    const float* W2   = (const float*)d_in[10];
    const float* b2   = (const float*)d_in[11];

    _Float16* Wih16 = (_Float16*)d_ws;                              // 2048*32 halves
    _Float16* Whh16 = (_Float16*)((char*)d_ws + (size_t)G4 * E_DIM * sizeof(_Float16));

    const int n_ih = G4 * E_DIM;    // 65536
    const int n_hh = G4 * H_DIM;    // 1048576
    f32_to_f16_kernel<<<(n_ih + 255) / 256, 256, 0, stream>>>(W_ih, Wih16, n_ih);
    f32_to_f16_kernel<<<(n_hh + 255) / 256, 256, 0, stream>>>(W_hh, Whh16, n_hh);

    controller_kernel<<<1, 512, 0, stream>>>(x, h0, c0, emb, b_ih, b_hh,
                                             W1, b1, W2, b2,
                                             Wih16, Whh16, (float*)d_out);
}
